// Encoder_Model_46325517254868
// MI455X (gfx1250) — matrix operand order repaired and checked by reading, not compiled, not run on hardware
//
#include <hip/hip_runtime.h>
#include <hip/hip_bf16.h>
#include <math.h>

typedef __attribute__((ext_vector_type(16))) _Float16 v16h;
typedef __attribute__((ext_vector_type(8)))  _Float16 v8h;
typedef __attribute__((ext_vector_type(4)))  _Float16 v4h;
typedef __attribute__((ext_vector_type(8)))  float    v8f;
typedef __attribute__((__vector_size__(16))) int      v4i;
typedef __attribute__((address_space(1))) v4i* gptr_v4i;
typedef __attribute__((address_space(3))) v4i* lptr_v4i;

#define NODE    30000
#define HID     128
#define NEDGE   600000
#define BATCHN  4096
#define NTASK   8192          // 4096 l-rows + 4096 r-rows
#define NTILES  1875          // 30000 / 16 column tiles
#define CSEG    8             // column segments (grid.y)
#define GAMMA_F 3.0f
#define LAMB_F  30.0f
#define TAU_F   10.0f

// ---------------- async global->LDS staging (CDNA5 path w/ fallback) ---------
#if defined(__has_builtin)
#if __has_builtin(__builtin_amdgcn_global_load_async_to_lds_b128)
#define ASYNC_COPY 1
#endif
#endif

__device__ inline void stage16B(const _Float16* gsrc, _Float16* ldst) {
#ifdef ASYNC_COPY
    __builtin_amdgcn_global_load_async_to_lds_b128(
        (gptr_v4i)(void*)gsrc, (lptr_v4i)(void*)ldst, 0, 0);
#else
    *(v8h*)ldst = *(const v8h*)gsrc;
#endif
}
__device__ inline void stage_wait() {
#ifdef ASYNC_COPY
    asm volatile("s_wait_asynccnt 0x0" ::: "memory");
#endif
}

// ---------------- scratch zeroing (feat + deg contiguous) ----------------
__global__ void zero_kernel(float* p, int n) {
    int i = blockIdx.x * blockDim.x + threadIdx.x;
    if (i < n) p[i] = 0.0f;
}

// ---------------- degree via global f32 atomics ----------------
__global__ void deg_kernel(const int* rows, float* deg) {
    int e = blockIdx.x * blockDim.x + threadIdx.x;
    if (e < NEDGE) atomicAdd(&deg[rows[e]], 1.0f);
}

// ---------------- feat[row] += (1/deg[row]) * emb[col] ----------------
__global__ void agg_kernel(const int* rows, const int* cols, const float* emb,
                           const float* deg, float* feat) {
    long long t = (long long)blockIdx.x * blockDim.x + threadIdx.x;
    int e = (int)(t >> 5);
    int lane = (int)(t & 31);
    if (e >= NEDGE) return;
    int r = rows[e], c = cols[e];
    float w = 1.0f / deg[r];
    int k0 = lane * 4;
    const float4 v = *(const float4*)(emb + (long long)c * HID + k0);
    float* dst = feat + (long long)r * HID + k0;
    atomicAdd(dst + 0, w * v.x);
    atomicAdd(dst + 1, w * v.y);
    atomicAdd(dst + 2, w * v.z);
    atomicAdd(dst + 3, w * v.w);
}

// ---------------- sqn + f16 conversion (one wave per node) ----------------
__global__ __launch_bounds__(256) void cvt_kernel(const float* feat, float* sqn,
                                                  _Float16* featH) {
    int wave = threadIdx.x >> 5;
    int lane = threadIdx.x & 31;
    int i = blockIdx.x * 8 + wave;
    if (i >= NODE) return;
    int k0 = lane * 4;
    const float4 f = *(const float4*)(feat + (long long)i * HID + k0);
    float s = f.x * f.x + f.y * f.y + f.z * f.z + f.w * f.w;
    for (int off = 16; off > 0; off >>= 1) s += __shfl_down(s, off);
    if (lane == 0) sqn[i] = s;
    v4h h; h[0] = (_Float16)f.x; h[1] = (_Float16)f.y;
    h[2] = (_Float16)f.z; h[3] = (_Float16)f.w;
    *(v4h*)(featH + (long long)i * HID + k0) = h;
}

// ---------------- pos_dis (one wave per pair) ----------------
__global__ __launch_bounds__(256) void pos_kernel(const int* tp, const float* feat,
                                                  float* pos) {
    int wave = threadIdx.x >> 5;
    int lane = threadIdx.x & 31;
    int b = blockIdx.x * 8 + wave;
    if (b >= BATCHN) return;
    int l = tp[2 * b + 0], r = tp[2 * b + 1];
    int k0 = lane * 4;
    const float4 fl = *(const float4*)(feat + (long long)l * HID + k0);
    const float4 fr = *(const float4*)(feat + (long long)r * HID + k0);
    float dx = fl.x - fr.x, dy = fl.y - fr.y, dz = fl.z - fr.z, dw = fl.w - fr.w;
    float s = dx * dx + dy * dy + dz * dz + dw * dw;
    for (int off = 16; off > 0; off >>= 1) s += __shfl_down(s, off);
    if (lane == 0) pos[b] = s;
}

// ---------------- fused distance/loss passes ----------------
// grid (64, 8): blockIdx.x = row group (128 rows, 16 per wave),
//               blockIdx.y = column segment (~234 of the 1875 tiles).
// B tiles (16 cols x 128 K = 4KB f16) are async-staged to LDS once per block
// and shared by all 8 waves; per-row stats reduced with wave32 shuffles.
template <int PASS>
__global__ __launch_bounds__(256) void pass_kernel(
    const int* tp, const float* sqn, const float* pos, const _Float16* featH,
    float* sumP, float* sqP, float* maxP,           // PASS 1 outputs
    const float* scaleR, const float* maxlR,        // PASS 2 inputs
    float* expP)                                    // PASS 2 output
{
    __shared__ __attribute__((aligned(64))) _Float16 sB[2][16 * HID];

    const int g = blockIdx.x;
    const int cs = blockIdx.y;
    const int tid = threadIdx.x;
    const int wave = tid >> 5;
    const int lane = tid & 31;
    const int n = lane & 15;
    const int khalf = lane >> 4;
    const int t0 = (cs * NTILES) / CSEG;
    const int t1 = ((cs + 1) * NTILES) / CSEG;
    const int rowbase = g * 128 + wave * 16;

    // ---- A fragments (16-bit 16x32, contiguous k = 16*khalf + s): row m = n of this wave ----
    int taskA = rowbase + n;
    int bA = taskA & (BATCHN - 1);
    int sideA = taskA >> 12;
    int nodeA = tp[2 * bA + sideA];
    v16h afrag[4];
    #pragma unroll
    for (int kt = 0; kt < 4; ++kt) {
        const _Float16* p = featH + (long long)nodeA * HID + kt * 32 + 16 * khalf;
        v8h lo = *(const v8h*)p;
        v8h hi = *(const v8h*)(p + 8);
        v16h a;
        #pragma unroll
        for (int i = 0; i < 8; ++i) { a[i] = lo[i]; a[8 + i] = hi[i]; }
        afrag[kt] = a;
    }

    // ---- per-lane row constants for its 8 C rows (m = v + 8*khalf) ----
    float rowc[8], scl[8], mx[8]; int lbv[8], rbv[8];
    #pragma unroll
    for (int v = 0; v < 8; ++v) {
        int m = v + 8 * khalf;
        int task = rowbase + m;
        int b = task & (BATCHN - 1);
        int side = task >> 12;
        int an = tp[2 * b + side];
        lbv[v] = tp[2 * b + 0];
        rbv[v] = tp[2 * b + 1];
        rowc[v] = pos[b] + GAMMA_F - sqn[an];
        if (PASS == 2) { scl[v] = scaleR[task]; mx[v] = maxlR[task]; }
    }

    float a0[8], a1[8], a2[8];
    #pragma unroll
    for (int v = 0; v < 8; ++v) { a0[v] = 0.0f; a1[v] = 0.0f; a2[v] = -1.0e30f; }

    const int colS = tid >> 4;      // staging: column 0..15
    const int chkS = tid & 15;      // staging: 8-half chunk 0..15
    // prologue: stage first tile
    stage16B(featH + (long long)(t0 * 16 + colS) * HID + chkS * 8,
             &sB[t0 & 1][colS * HID + chkS * 8]);

    for (int t = t0; t < t1; ++t) {
        stage_wait();
        __syncthreads();                         // buffer t&1 ready for all waves
        if (t + 1 < t1)
            stage16B(featH + (long long)((t + 1) * 16 + colS) * HID + chkS * 8,
                     &sB[(t + 1) & 1][colS * HID + chkS * 8]);

        // B (32x16 KxN, contiguous k order): lane n = column, khalf selects K 0-15/16-31
        const _Float16* bbase = &sB[t & 1][n * HID + 16 * khalf];
        v8f c = {};
        #pragma unroll
        for (int kt = 0; kt < 4; ++kt) {
            v16h bfrag = *(const v16h*)(bbase + kt * 32);
            c = __builtin_amdgcn_wmma_f32_16x16x32_f16(
                    false, afrag[kt], false, bfrag, (short)0, c, false, false);
        }

        int j = t * 16 + n;
        float sj = sqn[j];
        #pragma unroll
        for (int v = 0; v < 8; ++v) {
            float mask = 1.0f - (j == lbv[v] ? 1.0f : 0.0f)
                              - (j == rbv[v] ? 1.0f : 0.0f);
            float loss = (rowc[v] - sj + 2.0f * c[v]) * mask;
            if (PASS == 1) {
                a0[v] += loss;
                a1[v] += loss * loss;
                a2[v]  = fmaxf(a2[v], loss);
            } else {
                a0[v] += __expf(scl[v] * (loss - mx[v]));
            }
        }
        __syncthreads();                         // all waves done with buffer t&1
    }

    // ---- reduce across the 16 lanes of each khalf group (wave32 shuffles) ----
    #pragma unroll
    for (int off = 1; off < 16; off <<= 1) {
        #pragma unroll
        for (int v = 0; v < 8; ++v) {
            a0[v] += __shfl_xor(a0[v], off);
            if (PASS == 1) {
                a1[v] += __shfl_xor(a1[v], off);
                a2[v]  = fmaxf(a2[v], __shfl_xor(a2[v], off));
            }
        }
    }
    if (n == 0) {   // lanes 0 and 16 hold full sums for rows 0-7 / 8-15
        #pragma unroll
        for (int v = 0; v < 8; ++v) {
            int task = rowbase + v + 8 * khalf;
            if (PASS == 1) {
                sumP[task * CSEG + cs] = a0[v];
                sqP [task * CSEG + cs] = a1[v];
                maxP[task * CSEG + cs] = a2[v];
            } else {
                expP[task * CSEG + cs] = a0[v];
            }
        }
    }
}

// ---------------- combine per-segment stats -> scale / max / base ----------
__global__ void stats_kernel(const float* sumP, const float* sqP, const float* maxP,
                             float* scaleR, float* maxlR, float* baseR) {
    int r = blockIdx.x * blockDim.x + threadIdx.x;
    if (r >= NTASK) return;
    float s = 0.0f, q = 0.0f, m = -1.0e30f;
    #pragma unroll
    for (int cs = 0; cs < CSEG; ++cs) {
        s += sumP[r * CSEG + cs];
        q += sqP [r * CSEG + cs];
        m  = fmaxf(m, maxP[r * CSEG + cs]);
    }
    float mean = s * (1.0f / (float)NODE);
    float var = fmaxf(q * (1.0f / (float)NODE) - mean * mean, 1.0e-20f);
    float scale = LAMB_F * __frsqrt_rn(var);    // LAMB / std
    scaleR[r] = scale;
    maxlR[r] = m;
    baseR[r] = scale * (m - mean) + TAU_F;      // lse = base + log(sum exp)
}

// ---------------- final: lse per row, mean over batch ----------------
__global__ __launch_bounds__(256) void final_kernel(const float* expP,
                                                    const float* baseR, float* out) {
    __shared__ float s[256];
    float acc = 0.0f;
    for (int r = threadIdx.x; r < NTASK; r += 256) {
        float e = 0.0f;
        #pragma unroll
        for (int cs = 0; cs < CSEG; ++cs) e += expP[r * CSEG + cs];
        acc += baseR[r] + __logf(e);
    }
    s[threadIdx.x] = acc;
    __syncthreads();
    for (int off = 128; off > 0; off >>= 1) {
        if (threadIdx.x < off) s[threadIdx.x] += s[threadIdx.x + off];
        __syncthreads();
    }
    if (threadIdx.x == 0) out[0] = s[0] * (1.0f / (float)BATCHN);
}

extern "C" void kernel_launch(void* const* d_in, const int* in_sizes, int n_in,
                              void* d_out, int out_size, void* d_ws, size_t ws_size,
                              hipStream_t stream) {
    const int*   tp  = (const int*)d_in[0];     // (4096,2)
    const int*   adj = (const int*)d_in[1];     // (2,600000)
    const float* emb = (const float*)d_in[2];   // (30000,128)
    const int* rows = adj;
    const int* cols = adj + NEDGE;

    float* wsf     = (float*)d_ws;
    float* d_feat  = wsf;                        // 3,840,000 f32
    float* d_deg   = wsf + 3840000;              //    30,000 f32 (contig w/ feat)
    float* d_sqn   = wsf + 3870000;              //    30,000 f32
    float* d_pos   = wsf + 3900000;              //     4,096 f32
    _Float16* d_featH = (_Float16*)(wsf + 3904096); // 3,840,000 f16 (32B aligned)
    float* d_sumP  = wsf + 5824096;              // 65,536
    float* d_sqP   = wsf + 5889632;              // 65,536
    float* d_maxP  = wsf + 5955168;              // 65,536
    float* d_expP  = wsf + 6020704;              // 65,536
    float* d_scale = wsf + 6086240;              //  8,192
    float* d_maxl  = wsf + 6094432;              //  8,192
    float* d_base  = wsf + 6102624;              //  8,192   (~24.5 MB total)

    zero_kernel<<<(3870000 + 255) / 256, 256, 0, stream>>>(d_feat, 3870000);
    deg_kernel<<<(NEDGE + 255) / 256, 256, 0, stream>>>(rows, d_deg);
    agg_kernel<<<(NEDGE * 32 + 255) / 256, 256, 0, stream>>>(rows, cols, emb,
                                                             d_deg, d_feat);
    cvt_kernel<<<NODE / 8, 256, 0, stream>>>(d_feat, d_sqn, d_featH);
    pos_kernel<<<BATCHN / 8, 256, 0, stream>>>(tp, d_feat, d_pos);

    dim3 pgrid(NTASK / 128, CSEG);
    pass_kernel<1><<<pgrid, 256, 0, stream>>>(tp, d_sqn, d_pos, d_featH,
                                              d_sumP, d_sqP, d_maxP,
                                              nullptr, nullptr, nullptr);
    stats_kernel<<<NTASK / 256, 256, 0, stream>>>(d_sumP, d_sqP, d_maxP,
                                                  d_scale, d_maxl, d_base);
    pass_kernel<2><<<pgrid, 256, 0, stream>>>(tp, d_sqn, d_pos, d_featH,
                                              nullptr, nullptr, nullptr,
                                              d_scale, d_maxl, d_expP);
    final_kernel<<<1, 256, 0, stream>>>(d_expP, d_base, (float*)d_out);
}
